// HyperbolicGraphConv_58454504898751
// MI455X (gfx1250) — matrix operand order repaired and checked by reading, not compiled, not run on hardware
//
#include <hip/hip_runtime.h>
#include <math.h>

typedef __attribute__((ext_vector_type(16))) _Float16 v16h;
typedef __attribute__((ext_vector_type(8)))  float    v8f;

#define D    128      // feature dim (both in and out)
#define TM   64       // rows per block in the GEMM kernel
#define AP   132      // LDS pitch (halves) for A tile, even + padded
#define WP   132      // LDS pitch (halves) for transposed W tile

// ---------------------------------------------------------------------------
// Kernel 1: fused logmap0 + (x_tangent @ W + bias) via WMMA f16->f32.
// Block = 128 threads = 4 waves; each wave computes a 16x128 strip.
// ---------------------------------------------------------------------------
__global__ __launch_bounds__(128)
void hgc_logmap_gemm(const float* __restrict__ x,
                     const float* __restrict__ w,
                     const float* __restrict__ bias,
                     float* __restrict__ support,
                     int Ntot)
{
    __shared__ _Float16 Ah[TM * AP];   // logmap0(x) tile, row-major f16
    __shared__ _Float16 Wt[D * WP];    // W transposed: Wt[n][k] = w[k][n]

    const int t       = threadIdx.x;
    const int rowbase = blockIdx.x * TM;

    // ---- load weight (128x128) transposed into LDS as f16 ----
    for (int i = 0; i < D * D; i += 128) {
        int flat = i + t;
        int k = flat >> 7, n = flat & 127;
        Wt[n * WP + k] = (_Float16)w[flat];
    }

    // ---- logmap0: thread pair (t, t^1) handles one row (64 floats each) ----
    {
        int r = t >> 1;                 // local row 0..63
        int h = t & 1;                  // which half of the row
        int g = rowbase + r;            // global row
        float vbuf[64];
        float ss = 0.f;
        const float* src = x + (size_t)g * D + h * 64;
        #pragma unroll
        for (int j = 0; j < 64; ++j) {
            float v = (g < Ntot) ? src[j] : 0.f;
            vbuf[j] = v;
            ss += v * v;
        }
        ss += __shfl_xor(ss, 1);        // combine the two half-row sums
        float norm = sqrtf(ss);
        float nc   = fmaxf(norm, 1e-15f);
        float z    = fminf(nc, 1.f - 1e-5f);     // artanh domain clamp
        float s    = atanhf(z) / nc;             // logmap0 scale
        _Float16* dst = &Ah[r * AP + h * 64];
        #pragma unroll
        for (int j = 0; j < 64; ++j) dst[j] = (_Float16)(vbuf[j] * s);
    }
    __syncthreads();

    // ---- WMMA compute: wave wv owns rows [wv*16, wv*16+16) of the tile ----
    const int lane = t & 31;
    const int wv   = t >> 5;        // 0..3
    const int l16  = lane & 15;
    const int hf   = lane >> 4;     // lane half (A/B K-range selector)

    v8f acc[8];
    #pragma unroll
    for (int n = 0; n < 8; ++n)
        #pragma unroll
        for (int i = 0; i < 8; ++i) acc[n][i] = 0.f;

    #pragma unroll
    for (int ks = 0; ks < 4; ++ks) {            // K steps of 32
        // A fragment: 16x32 f16. Lane half hf holds K = k0+hf*8..+7 (V0..3)
        // and K = k0+16+hf*8..+7 (V4..7), 2 K-values (one u32) per VGPR.
        union { unsigned u[8]; v16h h; } a;
        const _Float16* ap = &Ah[(wv * 16 + l16) * AP];
        int kb = ks * 32 + hf * 8;
        #pragma unroll
        for (int j = 0; j < 4; ++j) {
            a.u[j]     = *(const unsigned*)&ap[kb + 2 * j];
            a.u[4 + j] = *(const unsigned*)&ap[kb + 16 + 2 * j];
        }
        #pragma unroll
        for (int n = 0; n < 8; ++n) {           // 8 N-tiles of 16
            // B fragment: 32x16 f16, column n = l16, lane half picks K range.
            union { unsigned u[8]; v16h h; } b;
            const _Float16* bp = &Wt[(n * 16 + l16) * WP + ks * 32 + hf * 8];
            #pragma unroll
            for (int j = 0; j < 4; ++j) {
                b.u[j]     = *(const unsigned*)&bp[2 * j];
                b.u[4 + j] = *(const unsigned*)&bp[16 + 2 * j];
            }
            acc[n] = __builtin_amdgcn_wmma_f32_16x16x32_f16(
                false, a.h, false, b.h, (short)0, acc[n], false, false);
        }
    }

    // ---- epilogue: C layout => VGPR v: M = hf*8 + v, N = l16 ----
    #pragma unroll
    for (int n = 0; n < 8; ++n) {
        int col = n * 16 + l16;
        float bv = bias[col];
        #pragma unroll
        for (int v = 0; v < 8; ++v) {
            int gr = rowbase + wv * 16 + hf * 8 + v;
            if (gr < Ntot)
                support[(size_t)gr * D + col] = acc[n][v] + bv;
        }
    }
}

// ---------------------------------------------------------------------------
// Kernel 2: zero the accumulator (d_out is poisoned by the harness).
// ---------------------------------------------------------------------------
__global__ __launch_bounds__(256)
void hgc_zero(float* __restrict__ p, long long n)
{
    long long i = (long long)blockIdx.x * blockDim.x + threadIdx.x;
    if (i < n) p[i] = 0.f;
}

// ---------------------------------------------------------------------------
// Kernel 3: SpMM scatter-add. One wave32 per edge; lane covers 4 features.
// support (102 MB) and out (102 MB) are L2-resident -> gathers/atomics hit L2.
// ---------------------------------------------------------------------------
__global__ __launch_bounds__(256)
void hgc_spmm(const float* __restrict__ support,
              const int* __restrict__ rows,
              const int* __restrict__ cols,
              const float* __restrict__ vals,
              float* __restrict__ out,
              int E)
{
    long long wid = ((long long)blockIdx.x * blockDim.x + threadIdx.x) >> 5;
    int lane = threadIdx.x & 31;
    if (wid >= E) return;
    int   r = rows[wid];
    int   c = cols[wid];
    float v = vals[wid];
    const float4 s = *(const float4*)(support + (size_t)c * D + lane * 4);
    float* o = out + (size_t)r * D + lane * 4;
    atomicAdd(o + 0, v * s.x);
    atomicAdd(o + 1, v * s.y);
    atomicAdd(o + 2, v * s.z);
    atomicAdd(o + 3, v * s.w);
}

// ---------------------------------------------------------------------------
// Kernel 4: expmap0 in place on d_out. One wave32 per row.
// ---------------------------------------------------------------------------
__global__ __launch_bounds__(256)
void hgc_expmap(float* __restrict__ out, int Ntot)
{
    int row  = (int)(((long long)blockIdx.x * blockDim.x + threadIdx.x) >> 5);
    int lane = threadIdx.x & 31;
    if (row >= Ntot) return;
    float* p = out + (size_t)row * D + lane * 4;
    float4 u = *(float4*)p;
    float ss = u.x * u.x + u.y * u.y + u.z * u.z + u.w * u.w;
    #pragma unroll
    for (int off = 16; off; off >>= 1) ss += __shfl_xor(ss, off);
    float norm = sqrtf(ss);
    float nc   = fmaxf(norm, 1e-15f);
    float s    = tanhf(nc) / nc;
    u.x *= s; u.y *= s; u.z *= s; u.w *= s;
    *(float4*)p = u;
}

// ---------------------------------------------------------------------------
extern "C" void kernel_launch(void* const* d_in, const int* in_sizes, int n_in,
                              void* d_out, int out_size, void* d_ws, size_t ws_size,
                              hipStream_t stream)
{
    const float* x    = (const float*)d_in[0];   // [N,128]
    const float* w    = (const float*)d_in[1];   // [128,128]
    const float* bias = (const float*)d_in[2];   // [128]
    const int*   rows = (const int*)d_in[3];     // [E]
    const int*   cols = (const int*)d_in[4];     // [E]
    const float* vals = (const float*)d_in[5];   // [E]
    float*       out  = (float*)d_out;           // [N,128]

    const int N = in_sizes[0] / D;               // 200000
    const int E = in_sizes[3];                   // 6400000

    float* support = (float*)d_ws;               // [N,128] f32 scratch

    // 1) logmap0 + GEMM + bias -> support
    hgc_logmap_gemm<<<(N + TM - 1) / TM, 128, 0, stream>>>(x, w, bias, support, N);

    // 2) zero accumulator
    long long total = (long long)N * D;
    hgc_zero<<<(unsigned)((total + 255) / 256), 256, 0, stream>>>(out, total);

    // 3) edge scatter-add (wave per edge, 8 edges per 256-thread block)
    unsigned spblocks = (unsigned)(((long long)E + 7) / 8);
    hgc_spmm<<<spblocks, 256, 0, stream>>>(support, rows, cols, vals, out, E);

    // 4) expmap0 in place (wave per row)
    unsigned exblocks = (unsigned)(((long long)N * 32 + 255) / 256);
    hgc_expmap<<<exblocks, 256, 0, stream>>>(out, N);
}
